// PinGUnet_new_87050397155922
// MI455X (gfx1250) — matrix operand order repaired and checked by reading, not compiled, not run on hardware
//
#include <hip/hip_runtime.h>

// ---------------------------------------------------------------------------
// PinG-UNet for MI455X (gfx1250, wave32, WMMA).
// Convs -> implicit GEMM with v_wmma_f32_16x16x32_f16 (f16 in, f32 acc).
//   - k_wmma_gemm      : 1 wave, 32M x 16N (2 acc, B frag reused)   [N%16]
//   - k_wmma_gemm_lds  : 4 waves, 32M x 64N, A strip double-buffered in LDS
//                        via GLOBAL_LOAD_ASYNC_TO_LDS_B128 + s_wait_asynccnt
// GNN   -> fused gather/MLP/scatter (HBM-gather bound; LDS-staged weights).
// ---------------------------------------------------------------------------

#define BATCH   2
#define HDIM    128
#define WDIM    128
#define HWDIM   (HDIM * WDIM)          // 16384
#define G_GRAPH (BATCH * HWDIM)        // 32768
#define N_NODES (4 * G_GRAPH)          // 131072
#define E_EDGES (4 * N_NODES)          // 524288

typedef __attribute__((ext_vector_type(16))) _Float16      v16h;
typedef __attribute__((ext_vector_type(8)))  float         v8f;
typedef __attribute__((ext_vector_type(4)))  unsigned int  v4u;

union F16Frag { v16h v; v4u q[2]; };

__device__ __forceinline__ v8f wmma_f16(const F16Frag& a, const F16Frag& b, v8f c) {
  return __builtin_amdgcn_wmma_f32_16x16x32_f16(
      /*neg_a=*/false, a.v, /*neg_b=*/false, b.v,
      /*c_mod=*/(short)0, c, /*reuse_a=*/false, /*reuse_b=*/false);
}

// C/D layout: VGPR v -> M = v + 8*kh, N col = lane&15.
__device__ __forceinline__ void store_tile(float* __restrict__ C, v8f acc,
                                           int mbase, int n, int kh, int N,
                                           int relu, int nchw, int OHW, float bv) {
#pragma unroll
  for (int v = 0; v < 8; ++v) {
    int m = mbase + v + kh * 8;
    float val = acc[v] + bv;
    if (relu) val = fmaxf(val, 0.0f);
    if (nchw) {
      int bb  = m / OHW;
      int pix = m - bb * OHW;
      C[((size_t)bb * N + n) * OHW + pix] = val;
    } else {
      C[(size_t)m * N + n] = val;
    }
  }
}

// ---------------------------------------------------------------------------
// v1: one wave per 32M x 16N.  A fragment per ISA 16-bit A 16x32 layout:
//   lane r=lane&15, kh=lane>>4; halves [k0+kh*8,+8) and [k0+16+kh*8,+8).
// Bt packed [N,K] row-major -> B fragment is two contiguous 16B loads.
// Requires M%32==0, N%16==0, K%32==0.
// ---------------------------------------------------------------------------
__global__ __launch_bounds__(32)
void k_wmma_gemm(const _Float16* __restrict__ A,
                 const _Float16* __restrict__ Bt,
                 const float* __restrict__ bias,
                 float* __restrict__ C,
                 int M, int N, int K,
                 int relu, int nchw, int OHW) {
  const int tileN = N >> 4;
  const int tid   = blockIdx.x;
  const int ti    = tid / tileN;        // 32-row strip
  const int tj    = tid - ti * tileN;   // 16-col tile
  const int lane  = threadIdx.x & 31;
  const int r     = lane & 15;
  const int kh    = lane >> 4;

  const _Float16* __restrict__ arow0 = A + (size_t)(ti * 32 + r) * K + kh * 8;
  const _Float16* __restrict__ arow1 = arow0 + (size_t)16 * K;
  const _Float16* __restrict__ brow  = Bt + (size_t)(tj * 16 + r) * K + kh * 8;

  v8f acc0 = {}, acc1 = {};
  for (int k0 = 0; k0 < K; k0 += 32) {
    __builtin_prefetch((const void*)(arow0 + k0 + 256), 0, 1);
    __builtin_prefetch((const void*)(arow1 + k0 + 256), 0, 1);
    F16Frag a0, a1, b;
    a0.q[0] = *(const v4u*)(arow0 + k0);
    a0.q[1] = *(const v4u*)(arow0 + k0 + 16);
    a1.q[0] = *(const v4u*)(arow1 + k0);
    a1.q[1] = *(const v4u*)(arow1 + k0 + 16);
    b.q[0]  = *(const v4u*)(brow + k0);
    b.q[1]  = *(const v4u*)(brow + k0 + 16);
    acc0 = wmma_f16(a0, b, acc0);
    acc1 = wmma_f16(a1, b, acc1);
  }
  const int n = tj * 16 + r;
  const float bv = bias ? bias[n] : 0.0f;
  store_tile(C, acc0, ti * 32,      n, kh, N, relu, nchw, OHW, bv);
  store_tile(C, acc1, ti * 32 + 16, n, kh, N, relu, nchw, OHW, bv);
}

// ---------------------------------------------------------------------------
// v2: 4 waves per block, 32M x 64N.  The 32-row A strip is shared by all 4
// waves: 64-K-half chunks are staged into LDS with async copies
// (GLOBAL_LOAD_ASYNC_TO_LDS_B128, ASYNCcnt) and double-buffered.
// Requires M%32==0, N%64==0, K%64==0.
// ---------------------------------------------------------------------------
__global__ __launch_bounds__(128)
void k_wmma_gemm_lds(const _Float16* __restrict__ A,
                     const _Float16* __restrict__ Bt,
                     const float* __restrict__ bias,
                     float* __restrict__ C,
                     int M, int N, int K,
                     int relu, int nchw, int OHW) {
  __shared__ _Float16 shA[2][32 * 64];   // 2 x 4KB chunks
  const int tiles64 = N >> 6;
  const int bid  = blockIdx.x;
  const int ti   = bid / tiles64;
  const int tj   = (bid - ti * tiles64) * 4 + (threadIdx.x >> 5);  // N tile per wave
  const int lane = threadIdx.x & 31;
  const int r    = lane & 15;
  const int kh   = lane >> 4;

  // async staging: thread t copies 2x16B of row (t>>2), segments (t&3)*2, +1
  const int srow = threadIdx.x >> 2;
  const int sseg = (threadIdx.x & 3) * 2;
  const _Float16* __restrict__ aBase = A + (size_t)(ti * 32 + srow) * K;
  const unsigned ldsBase = (unsigned)(size_t)(&shA[0][0]);

  const _Float16* __restrict__ brow = Bt + (size_t)(tj * 16 + r) * K + kh * 8;

  v8f acc0 = {}, acc1 = {};

  // prologue: stage chunk 0 into buffer 0
  {
    unsigned l0 = ldsBase + (unsigned)(srow * 128 + sseg * 16);
    unsigned long long g0 = (unsigned long long)(size_t)(aBase + sseg * 8);
    asm volatile("global_load_async_to_lds_b128 %0, %1, off" :: "v"(l0), "v"(g0) : "memory");
    asm volatile("global_load_async_to_lds_b128 %0, %1, off" :: "v"(l0 + 16u), "v"(g0 + 16ull) : "memory");
  }
  asm volatile("s_wait_asynccnt 0x0" ::: "memory");
  __syncthreads();

  const int nChunks = K >> 6;
  for (int c = 0; c < nChunks; ++c) {
    const int buf = c & 1;
    if (c + 1 < nChunks) {   // stage next chunk into the other buffer
      int k1 = (c + 1) << 6;
      unsigned l0 = ldsBase + (unsigned)((buf ^ 1) * (32 * 64 * 2) + srow * 128 + sseg * 16);
      unsigned long long g0 = (unsigned long long)(size_t)(aBase + k1 + sseg * 8);
      asm volatile("global_load_async_to_lds_b128 %0, %1, off" :: "v"(l0), "v"(g0) : "memory");
      asm volatile("global_load_async_to_lds_b128 %0, %1, off" :: "v"(l0 + 16u), "v"(g0 + 16ull) : "memory");
    }
    const int k0 = c << 6;
    const _Float16* sa = &shA[buf][0];
#pragma unroll
    for (int kk = 0; kk < 64; kk += 32) {
      F16Frag a0, a1, b;
      a0.q[0] = *(const v4u*)(sa + r * 64 + kk + kh * 8);
      a0.q[1] = *(const v4u*)(sa + r * 64 + kk + 16 + kh * 8);
      a1.q[0] = *(const v4u*)(sa + (16 + r) * 64 + kk + kh * 8);
      a1.q[1] = *(const v4u*)(sa + (16 + r) * 64 + kk + 16 + kh * 8);
      b.q[0]  = *(const v4u*)(brow + k0 + kk);
      b.q[1]  = *(const v4u*)(brow + k0 + kk + 16);
      acc0 = wmma_f16(a0, b, acc0);
      acc1 = wmma_f16(a1, b, acc1);
    }
    asm volatile("s_wait_asynccnt 0x0" ::: "memory");
    __syncthreads();
  }

  const int n = tj * 16 + r;
  const float bv = bias ? bias[n] : 0.0f;
  store_tile(C, acc0, ti * 32,      n, kh, N, relu, nchw, OHW, bv);
  store_tile(C, acc1, ti * 32 + 16, n, kh, N, relu, nchw, OHW, bv);
}

// im2col (NCHW f32 -> [B*OH*OW, Kpad] f16), k-order: k = cin*KHW*KHW + kh*KHW + kw
__global__ void k_im2col(const float* __restrict__ X, _Float16* __restrict__ col,
                         int Cin, int Hin, int Win, int OH, int OW,
                         int KHW, int stride, int pad, int K, int Kpad) {
  size_t idx = (size_t)blockIdx.x * blockDim.x + threadIdx.x;
  size_t total = (size_t)BATCH * OH * OW * Kpad;
  if (idx >= total) return;
  int    k = (int)(idx % Kpad);
  size_t m = idx / Kpad;
  float v = 0.0f;
  if (k < K) {
    int cin = k / (KHW * KHW);
    int t   = k - cin * KHW * KHW;
    int kkh = t / KHW, kkw = t - (t / KHW) * KHW;
    int ox = (int)(m % OW);
    int oy = (int)((m / OW) % OH);
    int bb = (int)(m / ((size_t)OW * OH));
    int iy = oy * stride - pad + kkh;
    int ix = ox * stride - pad + kkw;
    if (iy >= 0 && iy < Hin && ix >= 0 && ix < Win)
      v = X[(((size_t)bb * Cin + cin) * Hin + iy) * Win + ix];
  }
  col[idx] = (_Float16)v;
}

// Weight pack: W [Cout, K] f32 -> Bt [Cout, Kpad] f16 (zero-padded K)
__global__ void k_pack_w(const float* __restrict__ W, _Float16* __restrict__ Bt,
                         int Cout, int K, int Kpad) {
  size_t idx = (size_t)blockIdx.x * blockDim.x + threadIdx.x;
  size_t total = (size_t)Cout * Kpad;
  if (idx >= total) return;
  int k = (int)(idx % Kpad);
  int n = (int)(idx / Kpad);
  Bt[idx] = (_Float16)(k < K ? W[(size_t)n * K + k] : 0.0f);
}

// GroupNorm (+optional residual add, +optional ReLU).  grid = (groups, B)
__global__ __launch_bounds__(256)
void k_groupnorm(const float* __restrict__ X,
                 const float* __restrict__ gamma, const float* __restrict__ beta,
                 const float* __restrict__ res, float* __restrict__ Y,
                 int C, int HW, int groups, int relu) {
  int bb = blockIdx.y, g = blockIdx.x;
  int cpg = C / groups;
  size_t base = ((size_t)bb * C + (size_t)g * cpg) * HW;
  int nelem = cpg * HW;
  float s = 0.0f, ss = 0.0f;
  for (int i = threadIdx.x; i < nelem; i += blockDim.x) {
    float v = X[base + i]; s += v; ss += v * v;
  }
  __shared__ float sh0[256], sh1[256];
  sh0[threadIdx.x] = s; sh1[threadIdx.x] = ss;
  __syncthreads();
  for (int o = 128; o > 0; o >>= 1) {
    if ((int)threadIdx.x < o) {
      sh0[threadIdx.x] += sh0[threadIdx.x + o];
      sh1[threadIdx.x] += sh1[threadIdx.x + o];
    }
    __syncthreads();
  }
  float mu  = sh0[0] / (float)nelem;
  float var = sh1[0] / (float)nelem - mu * mu;
  float inv = rsqrtf(var + 1e-5f);
  for (int i = threadIdx.x; i < nelem; i += blockDim.x) {
    int c = g * cpg + i / HW;
    float v = (X[base + i] - mu) * inv * gamma[c] + beta[c];
    if (res)  v += res[base + i];
    if (relu) v = fmaxf(v, 0.0f);
    Y[base + i] = v;
  }
}

__global__ void k_zero(float* __restrict__ p, size_t n) {
  size_t i = (size_t)blockIdx.x * blockDim.x + threadIdx.x;
  if (i < n) p[i] = 0.0f;
}

// 2x nearest upsample into channel window [coff, coff+Cn) of a Ctot-channel dst
__global__ void k_up2(const float* __restrict__ src, float* __restrict__ dst,
                      int Cn, int hin, int win, int Ctot, int coff) {
  int h2 = hin * 2, w2 = win * 2;
  size_t idx = (size_t)blockIdx.x * blockDim.x + threadIdx.x;
  size_t total = (size_t)BATCH * Cn * h2 * w2;
  if (idx >= total) return;
  int x = (int)(idx % w2);
  int y = (int)((idx / w2) % h2);
  int c = (int)((idx / ((size_t)w2 * h2)) % Cn);
  int b = (int)(idx / ((size_t)w2 * h2 * Cn));
  dst[(((size_t)b * Ctot + coff + c) * h2 + y) * w2 + x] =
      src[(((size_t)b * Cn + c) * hin + (y >> 1)) * win + (x >> 1)];
}

// Copy channels into channel window [coff, coff+Cn) of a Ctot-channel dst
__global__ void k_copy_ch(const float* __restrict__ src, float* __restrict__ dst,
                          int Cn, int HW, int Ctot, int coff) {
  size_t idx = (size_t)blockIdx.x * blockDim.x + threadIdx.x;
  size_t total = (size_t)BATCH * Cn * HW;
  if (idx >= total) return;
  int pix = (int)(idx % HW);
  int c   = (int)((idx / HW) % Cn);
  int b   = (int)(idx / ((size_t)HW * Cn));
  dst[((size_t)b * Ctot + coff + c) * HW + pix] = src[idx];
}

// ---------------------------------------------------------------------------
// GNN: message MLP + scatter-add (fused).  m = [x[dst], x[src], e] @ We + be
// ---------------------------------------------------------------------------
__global__ __launch_bounds__(128)
void k_ping_msg(const float* __restrict__ h, int C,
                const float* __restrict__ ea,
                const int* __restrict__ srcI, const int* __restrict__ dstI,
                const float* __restrict__ We, const float* __restrict__ be,
                float* __restrict__ agg) {
  extern __shared__ float sW[];              // [Kin*16]
  __shared__ float sb[16];
  const int Kin = 2 * C + 4;
  for (int i = threadIdx.x; i < Kin * 16; i += blockDim.x) sW[i] = We[i];
  if (threadIdx.x < 16) sb[threadIdx.x] = be[threadIdx.x];
  __syncthreads();
  int e = blockIdx.x * blockDim.x + threadIdx.x;
  if (e >= E_EDGES) return;
  int d = dstI[e], s = srcI[e];
  float m[16];
#pragma unroll
  for (int j = 0; j < 16; ++j) m[j] = sb[j];
  const float* xd = h + (size_t)d * C;
  const float* xs = h + (size_t)s * C;
  for (int k = 0; k < C; ++k) {
    float v = xd[k]; const float* w = &sW[k * 16];
#pragma unroll
    for (int j = 0; j < 16; ++j) m[j] += v * w[j];
  }
  for (int k = 0; k < C; ++k) {
    float v = xs[k]; const float* w = &sW[(C + k) * 16];
#pragma unroll
    for (int j = 0; j < 16; ++j) m[j] += v * w[j];
  }
  for (int k = 0; k < 4; ++k) {
    float v = ea[(size_t)e * 4 + k]; const float* w = &sW[(2 * C + k) * 16];
#pragma unroll
    for (int j = 0; j < 16; ++j) m[j] += v * w[j];
  }
#pragma unroll
  for (int j = 0; j < 16; ++j) atomicAdd(&agg[(size_t)d * 16 + j], m[j]);
}

// out = [h, agg] @ Wo + bo  (Cout = 32), optional relu
__global__ __launch_bounds__(128)
void k_ping_out(const float* __restrict__ h, int C,
                const float* __restrict__ agg,
                const float* __restrict__ Wo, const float* __restrict__ bo,
                float* __restrict__ out, int relu) {
  extern __shared__ float sW[];              // [(C+16)*32]
  __shared__ float sb[32];
  const int Kin = C + 16;
  for (int i = threadIdx.x; i < Kin * 32; i += blockDim.x) sW[i] = Wo[i];
  if (threadIdx.x < 32) sb[threadIdx.x] = bo[threadIdx.x];
  __syncthreads();
  int n = blockIdx.x * blockDim.x + threadIdx.x;
  if (n >= N_NODES) return;
  float o[32];
#pragma unroll
  for (int j = 0; j < 32; ++j) o[j] = sb[j];
  const float* hp = h + (size_t)n * C;
  for (int k = 0; k < C; ++k) {
    float v = hp[k]; const float* w = &sW[k * 32];
#pragma unroll
    for (int j = 0; j < 32; ++j) o[j] += v * w[j];
  }
  const float* ap = agg + (size_t)n * 16;
  for (int k = 0; k < 16; ++k) {
    float v = ap[k]; const float* w = &sW[(C + k) * 32];
#pragma unroll
    for (int j = 0; j < 32; ++j) o[j] += v * w[j];
  }
#pragma unroll
  for (int j = 0; j < 32; ++j) {
    float v = o[j];
    if (relu) v = fmaxf(v, 0.0f);
    out[(size_t)n * 32 + j] = v;
  }
}

// gf[g, c] = mean over the 4 nodes of graph g (graph_index = arange(N)//4)
__global__ void k_graph_mean(const float* __restrict__ h, float* __restrict__ gf) {
  size_t idx = (size_t)blockIdx.x * blockDim.x + threadIdx.x;
  if (idx >= (size_t)G_GRAPH * 32) return;
  size_t g = idx / 32; int c = (int)(idx % 32);
  size_t b = (size_t)(4 * g) * 32 + c;
  gf[idx] = 0.25f * (h[b] + h[b + 32] + h[b + 64] + h[b + 96]);
}

// Build f16 rows [B*H*W, 64]: cols 0..31 = xs_rows, cols 32..63 = gf_img per
// gf.T.reshape(-1,32,H,W):  gf_img[b,c,y,x] = gf[(c&1)*HW + y*W + x, b*16 + c/2]
__global__ void k_build_scale_in(const float* __restrict__ xs_rows,
                                 const float* __restrict__ gf,
                                 _Float16* __restrict__ A2) {
  size_t idx = (size_t)blockIdx.x * blockDim.x + threadIdx.x;
  size_t total = (size_t)BATCH * HWDIM * 64;
  if (idx >= total) return;
  int    c   = (int)(idx % 64);
  size_t row = idx / 64;
  int pix = (int)(row % HWDIM);
  int b   = (int)(row / HWDIM);
  float v;
  if (c < 32) {
    v = xs_rows[row * 32 + c];
  } else {
    int cc  = c - 32;
    int gid = (cc & 1) * HWDIM + pix;
    int ch  = b * 16 + (cc >> 1);
    v = gf[(size_t)gid * 32 + ch];
  }
  A2[idx] = (_Float16)v;
}

// ---------------------------------------------------------------------------
// Host orchestration
// ---------------------------------------------------------------------------
static inline int ngroups_of(int c) { return (c % 32 != 0 || c == 32) ? c / 2 : 32; }
static inline int kpad64(int k) { return (k + 63) & ~63; }

// Input index map.  Top-level insertion order: x, node_x, edge_attr,
// params (pytree: dicts sorted by key, depth-first), edge_index, graph_index.
enum {
  IN_X = 0, IN_NODE_X = 1, IN_EDGE_ATTR = 2,
  P_DEC = 3,                 // 12 cgr blocks * 3 = 36 leaves -> 3..38
  P_ENC_CONV1 = 39,
  P_EN0 = 40,                // down(9) + plain(6)
  P_EN1 = 55, P_EN2 = 70, P_EN3 = 85, P_EN4 = 100,
  P_GN1_G = 115, P_GN1_B = 116,
  P_GSCALE_W = 117, P_GSCALE_B = 118,
  P_GNN0 = 119, P_GNN1 = 123, P_GNN2 = 127,   // each: We, Wo, be, bo
  P_SCALE_W = 131, P_SCALE_B = 132,
  IN_EDGE_INDEX = 133, IN_GRAPH_INDEX = 134,
};

extern "C" void kernel_launch(void* const* d_in, const int* in_sizes, int n_in,
                              void* d_out, int out_size, void* d_ws, size_t ws_size,
                              hipStream_t stream) {
  (void)in_sizes; (void)n_in; (void)out_size; (void)ws_size;

  auto fp = [&](int i) { return (const float*)d_in[i]; };

  const float* x_img   = fp(IN_X);
  const float* node_x  = fp(IN_NODE_X);
  const float* ea      = fp(IN_EDGE_ATTR);
  const int*   eidx    = (const int*)d_in[IN_EDGE_INDEX];
  const int*   srcI    = eidx;
  const int*   dstI    = eidx + E_EDGES;

  // ---- workspace bump allocator ----
  char* wsp = (char*)d_ws;
  auto allocb = [&](size_t bytes) {
    void* p = (void*)wsp;
    wsp += (bytes + 255) & ~(size_t)255;
    return p;
  };
  auto allocf = [&](size_t n) { return (float*)allocb(n * sizeof(float)); };
  auto alloch = [&](size_t n) { return (_Float16*)allocb(n * sizeof(_Float16)); };

  float*    h1      = allocf((size_t)N_NODES * 32);
  float*    h2      = allocf((size_t)N_NODES * 32);
  float*    agg     = allocf((size_t)N_NODES * 16);
  float*    gf      = allocf((size_t)G_GRAPH * 32);
  float*    xs_rows = allocf((size_t)BATCH * HWDIM * 32);
  _Float16* A2      = alloch((size_t)BATCH * HWDIM * 64);
  _Float16* colbuf  = alloch((size_t)10600000);       // max 32768*320 halves
  _Float16* wbuf    = alloch((size_t)2400000);        // max 512*4608 halves
  float*    in_x    = allocf((size_t)BATCH * 32 * HWDIM);
  const size_t TSZ  = 1100000;                        // >= 2*32*128*128
  float* t0 = allocf(TSZ); float* t1 = allocf(TSZ);
  float* t2 = allocf(TSZ); float* t3 = allocf(TSZ);
  float* t4 = allocf(TSZ); float* t5 = allocf(TSZ);
  float* dA = allocf(TSZ); float* dB = allocf(TSZ);
  float* feat0 = allocf((size_t)BATCH * 32  * 64 * 64);
  float* feat1 = allocf((size_t)BATCH * 64  * 32 * 32);
  float* feat2 = allocf((size_t)BATCH * 128 * 16 * 16);
  float* feat3 = allocf((size_t)BATCH * 256 * 8 * 8);
  float* feat4 = allocf((size_t)BATCH * 512 * 4 * 4);

  auto blocks = [](size_t n, int bs) { return (unsigned)((n + bs - 1) / bs); };

  // GEMM dispatch: v2 (async-LDS, 4 waves) when N%64==0, else v1.
  auto gemm = [&](const _Float16* A, const _Float16* Bt, const float* bias,
                  float* Y, int M, int N, int K, int relu, int nchw, int OHW) {
    if ((N & 63) == 0) {
      unsigned nb = (unsigned)(((size_t)M / 32) * (N / 64));
      k_wmma_gemm_lds<<<nb, 128, 0, stream>>>(A, Bt, bias, Y, M, N, K, relu, nchw, OHW);
    } else {
      unsigned nb = (unsigned)(((size_t)M / 32) * (N / 16));
      k_wmma_gemm<<<nb, 32, 0, stream>>>(A, Bt, bias, Y, M, N, K, relu, nchw, OHW);
    }
  };

  auto conv = [&](const float* X, int Cin, int Hin, int Wi, int Cout,
                  int k, int s, int p, float* Y) -> int {
    int OH = (Hin + 2 * p - k) / s + 1;
    int K  = Cin * k * k;
    int Kp = kpad64(K);
    size_t M = (size_t)BATCH * OH * OH;
    k_im2col<<<blocks(M * Kp, 256), 256, 0, stream>>>(X, colbuf, Cin, Hin, Hin,
                                                      OH, OH, k, s, p, K, Kp);
    k_pack_w<<<blocks((size_t)Cout * Kp, 256), 256, 0, stream>>>(fp(Wi), wbuf, Cout, K, Kp);
    gemm(colbuf, wbuf, nullptr, Y, (int)M, Cout, Kp, 0, 1, OH * OH);
    return OH;
  };

  auto gn = [&](const float* X, float* Y, int C, int HW, int groups,
                int gi, int bi, const float* res, int relu) {
    dim3 gg(groups, BATCH);
    k_groupnorm<<<gg, 256, 0, stream>>>(X, fp(gi), fp(bi), res, Y, C, HW, groups, relu);
  };

  // Down res:  base+{0:c1, 1:c2, 2:dg.g, 3:dg.b, 4:dw, 5:g1.g, 6:g1.b, 7:g2.g, 8:g2.b}
  // Plain res: base+{0:c1, 1:c2, 2:g1.g, 3:g1.b, 4:g2.g, 5:g2.b}
  auto resblock = [&](const float* X, int Cin, int Hin, int Cout, bool down,
                      int base, float* out) {
    int s = down ? 2 : 1;
    int c1 = base, c2 = base + 1;
    int g1g, g1b, g2g, g2b, dw = -1, dgg = -1, dgb = -1;
    if (down) { dgg = base + 2; dgb = base + 3; dw = base + 4;
                g1g = base + 5; g1b = base + 6; g2g = base + 7; g2b = base + 8; }
    else      { g1g = base + 2; g1b = base + 3; g2g = base + 4; g2b = base + 5; }
    int grp = ngroups_of(Cout);
    int OH = conv(X, Cin, Hin, c1, Cout, 3, s, 1, t2);
    gn(t2, t2, Cout, OH * OH, grp, g1g, g1b, nullptr, 1);
    conv(t2, Cout, OH, c2, Cout, 3, 1, 1, t3);
    const float* res;
    if (down) {
      conv(X, Cin, Hin, dw, Cout, 1, 2, 0, t4);
      gn(t4, t4, Cout, OH * OH, grp, dgg, dgb, nullptr, 0);
      res = t4;
    } else {
      res = X;
    }
    gn(t3, out, Cout, OH * OH, grp, g2g, g2b, res, 1);
  };

  auto dec_block = [&](const float* X, int Cx, const float* skip, int Cs,
                       int hin, bool up, int wi, int Cout, float* out) {
    int h = up ? 2 * hin : hin;
    int Ct = Cx + Cs;
    const float* cin = X;
    if (up) {
      size_t tot = (size_t)BATCH * Cx * h * h;
      k_up2<<<blocks(tot, 256), 256, 0, stream>>>(X, t5, Cx, hin, hin, Ct, 0);
      cin = t5;
    }
    if (skip) {
      size_t tot = (size_t)BATCH * Cs * h * h;
      k_copy_ch<<<blocks(tot, 256), 256, 0, stream>>>(skip, t5, Cs, h * h, Ct, Cx);
      cin = t5;
    }
    int grp = ngroups_of(Cout);
    conv(cin, Ct, h, wi, Cout, 3, 1, 1, t2);
    gn(t2, t2, Cout, h * h, grp, wi + 1, wi + 2, nullptr, 1);
    conv(t2, Cout, h, wi + 3, Cout, 3, 1, 1, t3);
    gn(t3, out, Cout, h * h, grp, wi + 4, wi + 5, nullptr, 1);
  };

  // ================== GNN ==================
  auto gnn_layer = [&](const float* hin, int C, int pbase, float* hout, int relu) {
    k_zero<<<blocks((size_t)N_NODES * 16, 256), 256, 0, stream>>>(agg, (size_t)N_NODES * 16);
    int Kin = 2 * C + 4;
    k_ping_msg<<<blocks(E_EDGES, 128), 128, (size_t)Kin * 16 * 4, stream>>>(
        hin, C, ea, srcI, dstI, fp(pbase + 0), fp(pbase + 2), agg);
    k_ping_out<<<blocks(N_NODES, 128), 128, (size_t)(C + 16) * 32 * 4, stream>>>(
        hin, C, agg, fp(pbase + 1), fp(pbase + 3), hout, relu);
  };
  gnn_layer(node_x, 16, P_GNN0, h1, 1);
  gnn_layer(h1,     32, P_GNN1, h2, 1);
  gnn_layer(h2,     32, P_GNN2, h1, 0);
  k_graph_mean<<<blocks((size_t)G_GRAPH * 32, 256), 256, 0, stream>>>(h1, gf);

  // ================== graph feature fuse ==================
  {
    // xs = conv1x1(x, g_scale) + b   -> row-major [B*HW, 32]
    int K = 8, Kp = 64;
    size_t M = (size_t)BATCH * HWDIM;
    k_im2col<<<blocks(M * Kp, 256), 256, 0, stream>>>(x_img, colbuf, 8, HDIM, WDIM,
                                                      HDIM, WDIM, 1, 1, 0, K, Kp);
    k_pack_w<<<blocks((size_t)32 * Kp, 256), 256, 0, stream>>>(fp(P_GSCALE_W), wbuf, 32, K, Kp);
    gemm(colbuf, wbuf, fp(P_GSCALE_B), xs_rows, (int)M, 32, Kp, 0, 0, 0);
    // in_x = conv1x1(concat(xs, gf_img), scale) + b  -> NCHW [2,32,128,128]
    k_build_scale_in<<<blocks(M * 64, 256), 256, 0, stream>>>(xs_rows, gf, A2);
    k_pack_w<<<blocks((size_t)32 * 64, 256), 256, 0, stream>>>(fp(P_SCALE_W), wbuf, 32, 64, 64);
    gemm(A2, wbuf, fp(P_SCALE_B), in_x, (int)M, 32, 64, 0, 1, HWDIM);
  }

  // ================== encoder ==================
  conv(in_x, 32, 128, P_ENC_CONV1, 16, 3, 1, 1, t0);
  gn(t0, t0, 16, HWDIM, 8, P_GN1_G, P_GN1_B, nullptr, 1);

  resblock(t0,    16, 128, 32,  true,  P_EN0,      t1);
  resblock(t1,    32, 64,  32,  false, P_EN0 + 9,  feat0);
  resblock(feat0, 32, 64,  64,  true,  P_EN1,      t1);
  resblock(t1,    64, 32,  64,  false, P_EN1 + 9,  feat1);
  resblock(feat1, 64, 32,  128, true,  P_EN2,      t1);
  resblock(t1,    128, 16, 128, false, P_EN2 + 9,  feat2);
  resblock(feat2, 128, 16, 256, true,  P_EN3,      t1);
  resblock(t1,    256, 8,  256, false, P_EN3 + 9,  feat3);
  resblock(feat3, 256, 8,  512, true,  P_EN4,      t1);
  resblock(t1,    512, 4,  512, false, P_EN4 + 9,  feat4);

  // ================== decoder ==================
  dec_block(feat4, 512, nullptr, 0,   4,  false, P_DEC + 0,  512, dA); // center
  dec_block(dA,    512, feat3,   256, 4,  true,  P_DEC + 6,  256, dB); // l1
  dec_block(dB,    256, feat2,   128, 8,  true,  P_DEC + 12, 128, dA); // l2
  dec_block(dA,    128, feat1,   64,  16, true,  P_DEC + 18, 64,  dB); // l3
  dec_block(dB,    64,  feat0,   32,  32, true,  P_DEC + 24, 32,  dA); // l4
  // l5: final GN(+ReLU) writes straight to d_out; resize(128->128) is identity
  dec_block(dA,    32,  nullptr, 0,   64, true,  P_DEC + 30, 32, (float*)d_out);
}
